// HetGATV2_17222818857489
// MI455X (gfx1250) — matrix operand order repaired and checked by reading, not compiled, not run on hardware
//
#include <hip/hip_runtime.h>

typedef __attribute__((ext_vector_type(16))) _Float16 v16h;
typedef __attribute__((ext_vector_type(8)))  float    v8f;

#define BD 4
#define UD 32
#define ADIM 64
#define HHEADS 40
#define PMAX 1.0f

// ----------------------------------------------------------------------------
// Elementwise / utility kernels
// ----------------------------------------------------------------------------
__global__ void fill_kernel(float* p, long n, float v) {
  long i = (long)blockIdx.x * blockDim.x + threadIdx.x;
  if (i < n) p[i] = v;
}

__global__ void add_kernel(const float* a, const float* b, float* o, long n) {
  long i = (long)blockIdx.x * blockDim.x + threadIdx.x;
  if (i < n) o[i] = a[i] + b[i];
}

// h[n,c] = (sum_f x[n,f]) * w[c] + F*b[c]
__global__ void embed_kernel(const float* x, int N, int F, const float* w,
                             const float* b, float* out) {
  int i = blockIdx.x * blockDim.x + threadIdx.x;
  if (i >= N * 128) return;
  int n = i >> 7, c = i & 127;
  float s = 0.f;
  for (int f = 0; f < F; ++f) s += x[n * F + f];
  out[i] = s * w[c] + (float)F * b[c];
}

// GraphNorm over rows (axis 0), one thread per channel (coalesced across c).
// act: 0=none, 1=relu, 2=elu. In-place.
__global__ void graph_norm_kernel(float* x, int N, int C, const float* g,
                                  const float* bb, const float* aa, int act) {
  int c = blockIdx.x * blockDim.x + threadIdx.x;
  if (c >= C) return;
  float m = 0.f;
  for (int n = 0; n < N; ++n) m += x[(long)n * C + c];
  m /= (float)N;
  float am = aa[c] * m;
  float v = 0.f;
  for (int n = 0; n < N; ++n) {
    float d = x[(long)n * C + c] - am;
    v += d * d;
  }
  v /= (float)N;
  float inv = rsqrtf(v + 1e-5f);
  float gg = g[c], bc = bb[c];
  for (int n = 0; n < N; ++n) {
    float y = gg * (x[(long)n * C + c] - am) * inv + bc;
    if (act == 1) y = fmaxf(y, 0.f);
    else if (act == 2) y = (y > 0.f) ? y : (expf(y) - 1.f);
    x[(long)n * C + c] = y;
  }
}

// ----------------------------------------------------------------------------
// WMMA GEMM: C[M,Ncols] (+)= A[M,K] @ B[K,Ncols] (+ bias)
//  - fp32 in memory, f16 fragments, f32 accumulate (v_wmma_f32_16x16x32_f16)
//  - optional row-gather on A via aidx (A row for output row i = A[aidx[i]])
//  - M multiple of 16, Ncols multiple of 32; K arbitrary (branch-free tail)
//  - one wave per 16x32 output tile (two 16x16 WMMA accumulators sharing the
//    A fragment); 8 waves per 256-thread block
// Fragment layouts (wave32; A/B in contiguous k order, C/D per CDNA5 ISA 7.12.2):
//  A 16x32 f16 : lane m=L&15; halves 0..7 -> k=kk+kb_b+h (contiguous),
//                halves 8..15 -> k=kk+kb_b+8+(h-8) (contiguous) => float4 x4
//  B 32x16 f16 : lane n=L&15; half h -> k = kk + h + ((L>>4)<<4)
//  C/D 16x16 f32: lane n=L&15; vgpr r -> m = r + 8*(L>>4)
// ----------------------------------------------------------------------------
__global__ void gemm_wmma_kernel(const float* __restrict__ Ap,
                                 const int* __restrict__ aidx,
                                 const float* __restrict__ Bp,
                                 const float* __restrict__ bias,
                                 float* __restrict__ Cp,
                                 int M, int N, int K, int lda, int acc) {
  int gtid = blockIdx.x * blockDim.x + threadIdx.x;
  int wave = gtid >> 5;
  int lane = threadIdx.x & 31;
  int Nt2 = N >> 5;  // 16x32 tiles along N
  int Mt = M >> 4;
  if (wave >= Mt * Nt2) return;  // wave-uniform: EXEC stays all-ones for WMMA
  int tm = wave / Nt2;
  int tn = wave - tm * Nt2;

  int mrow = tm * 16 + (lane & 15);
  long arow = aidx ? (long)aidx[mrow] : (long)mrow;
  const float* Arow = Ap + arow * (long)lda;
  int col0 = tn * 32 + (lane & 15);
  int col1 = col0 + 16;
  int hi = lane >> 4;
  int kb_a = hi << 3;   // 0 or 8
  int kb_b = hi << 4;   // 0 or 16

  v8f c0, c1;
  if (acc) {
#pragma unroll
    for (int r = 0; r < 8; ++r) {
      long mr = tm * 16 + r + 8 * hi;
      c0[r] = Cp[mr * N + col0];
      c1[r] = Cp[mr * N + col1];
    }
  } else {
#pragma unroll
    for (int r = 0; r < 8; ++r) { c0[r] = 0.f; c1[r] = 0.f; }
  }

  int Kmain = K & ~31;
  for (int kk = 0; kk < Kmain; kk += 32) {
    if (kk + 32 < Kmain)  // uniform branch; lowers to global_prefetch_b8
      __builtin_prefetch(Bp + (long)(kk + 32 + kb_b) * N + col0, 0, 1);
    // A fragment: 4x global_load_b128, no guards
    const float4* a0p = (const float4*)(Arow + kk + kb_b);
    const float4* a1p = (const float4*)(Arow + kk + kb_b + 8);
    float4 fa0 = a0p[0], fa1 = a0p[1];
    float4 fa2 = a1p[0], fa3 = a1p[1];
    v16h a;
    a[0]  = (_Float16)fa0.x; a[1]  = (_Float16)fa0.y;
    a[2]  = (_Float16)fa0.z; a[3]  = (_Float16)fa0.w;
    a[4]  = (_Float16)fa1.x; a[5]  = (_Float16)fa1.y;
    a[6]  = (_Float16)fa1.z; a[7]  = (_Float16)fa1.w;
    a[8]  = (_Float16)fa2.x; a[9]  = (_Float16)fa2.y;
    a[10] = (_Float16)fa2.z; a[11] = (_Float16)fa2.w;
    a[12] = (_Float16)fa3.x; a[13] = (_Float16)fa3.y;
    a[14] = (_Float16)fa3.z; a[15] = (_Float16)fa3.w;
    // B fragments: unguarded row-strided loads (cluster into one clause)
    const float* bp = Bp + (long)(kk + kb_b) * N;
    v16h b0, b1;
#pragma unroll
    for (int h = 0; h < 16; ++h) {
      b0[h] = (_Float16)bp[(long)h * N + col0];
      b1[h] = (_Float16)bp[(long)h * N + col1];
    }
    c0 = __builtin_amdgcn_wmma_f32_16x16x32_f16(false, a, false, b0,
                                                (short)0, c0, false, false);
    c1 = __builtin_amdgcn_wmma_f32_16x16x32_f16(false, a, false, b1,
                                                (short)0, c1, false, false);
  }

  if (Kmain < K) {  // branch-free guarded tail (clamped address + cndmask)
    int kk = Kmain;
    v16h a, b0, b1;
#pragma unroll
    for (int h = 0; h < 16; ++h) {
      int ka = kk + h + kb_b;
      bool oka = ka < K;
      float va = Arow[oka ? ka : 0];
      a[h] = (_Float16)(oka ? va : 0.f);
      int kb = kk + h + kb_b;
      bool okb = kb < K;
      long krow = (long)(okb ? kb : 0) * N;
      float vb0 = Bp[krow + col0];
      float vb1 = Bp[krow + col1];
      b0[h] = (_Float16)(okb ? vb0 : 0.f);
      b1[h] = (_Float16)(okb ? vb1 : 0.f);
    }
    c0 = __builtin_amdgcn_wmma_f32_16x16x32_f16(false, a, false, b0,
                                                (short)0, c0, false, false);
    c1 = __builtin_amdgcn_wmma_f32_16x16x32_f16(false, a, false, b1,
                                                (short)0, c1, false, false);
  }

  float bv0 = bias ? bias[col0] : 0.f;
  float bv1 = bias ? bias[col1] : 0.f;
#pragma unroll
  for (int r = 0; r < 8; ++r) {
    long mr = tm * 16 + r + 8 * hi;
    Cp[mr * N + col0] = c0[r] + bv0;
    Cp[mr * N + col1] = c1[r] + bv1;
  }
}

// ----------------------------------------------------------------------------
// GATv2 attention pieces
// ----------------------------------------------------------------------------
__global__ void edge_logit_kernel(const float* q, const float* k, const float* ef,
                                  const int* src, const int* dst, const float* att,
                                  float* logit, int E, int dh, int hd) {
  int i = blockIdx.x * blockDim.x + threadIdx.x;
  if (i >= E * HHEADS) return;
  int e = i / HHEADS, h = i - e * HHEADS;
  int s = src[e], d = dst[e];
  const float* qp = q + (long)d * hd + h * dh;
  const float* kp = k + (long)s * hd + h * dh;
  const float* ep = ef + (long)e * hd + h * dh;
  const float* ap = att + h * dh;
  float acc = 0.f;
  for (int t = 0; t < dh; ++t) {
    float z = qp[t] + kp[t] + ep[t];
    z = (z > 0.f) ? z : 0.2f * z;
    acc += z * ap[t];
  }
  logit[(long)e * HHEADS + h] = acc;
}

__device__ __forceinline__ long seg_base(int n, int perB, int epb, int baseMul) {
  return (long)(n / perB) * epb + (long)(n % perB) * baseMul;
}

__global__ void seg_softmax_kernel(const float* logit, float* alpha, int Ndst,
                                   int deg, int stride, int perB, int epb,
                                   int baseMul) {
  int i = blockIdx.x * blockDim.x + threadIdx.x;
  if (i >= Ndst * HHEADS) return;
  int n = i / HHEADS, h = i - n * HHEADS;
  long base = seg_base(n, perB, epb, baseMul);
  float m = -3.0e38f;
  for (int t = 0; t < deg; ++t)
    m = fmaxf(m, logit[(base + (long)t * stride) * HHEADS + h]);
  float s = 0.f;
  for (int t = 0; t < deg; ++t)
    s += expf(logit[(base + (long)t * stride) * HHEADS + h] - m);
  float invs = 1.f / (s + 1e-16f);
  for (int t = 0; t < deg; ++t) {
    long e = base + (long)t * stride;
    alpha[e * HHEADS + h] = expf(logit[e * HHEADS + h] - m) * invs;
  }
}

// out[n, c] += sum_t alpha[e,h] * (v[src[e], c] + ef[e, c]),  h = c/dh
__global__ void aggregate_kernel(const float* v, const float* ef,
                                 const float* alpha, const int* src, float* out,
                                 int Ndst, int hd, int dh, int deg, int stride,
                                 int perB, int epb, int baseMul) {
  long i = (long)blockIdx.x * blockDim.x + threadIdx.x;
  if (i >= (long)Ndst * hd) return;
  int n = (int)(i / hd), c = (int)(i - (long)n * hd);
  int h = c / dh;
  long base = seg_base(n, perB, epb, baseMul);
  float acc = 0.f;
  for (int t = 0; t < deg; ++t) {
    long e = base + (long)t * stride;
    int s = src[e];
    acc += alpha[e * HHEADS + h] * (v[(long)s * hd + c] + ef[e * hd + c]);
  }
  out[i] += acc;
}

// Tiny output projections (N = 1 or 2): scalar dot products.
__global__ void gemv_small_kernel(const float* A, const float* B,
                                  const float* bias, float* C, int M, int N,
                                  int K) {
  int i = blockIdx.x * blockDim.x + threadIdx.x;
  if (i >= M * N) return;
  int m = i / N, n = i - m * N;
  float acc = bias ? bias[n] : 0.f;
  const float* ar = A + (long)m * K;
  for (int k = 0; k < K; ++k) acc += ar[k] * B[(long)k * N + n];
  C[i] = acc;
}

// Final beamformer: RFu = RF/|RF|; F = BB@RFu; row-normalize; W = sqrt(pw)*F.
__global__ void precoder_kernel(const float* P, const float* bbo,
                                const float* rfo, float* out) {
  int b = blockIdx.x / UD;
  int u = blockIdx.x - b * UD;
  int a = threadIdx.x;
  float pm = -3.0e38f;
  for (int t = 0; t < UD; ++t) pm = fmaxf(pm, P[b * UD + t]);
  float ps = 0.f;
  for (int t = 0; t < UD; ++t) ps += expf(P[b * UD + t] - pm);
  float pw = PMAX * expf(P[b * UD + u] - pm) / ps;
  float amp = sqrtf(pw);

  float Fr = 0.f, Fi = 0.f;
  for (int j = 0; j < UD; ++j) {
    const float* bp = bbo + ((long)(b * UD + u) * UD + j) * 2;
    float br = bp[0], bi = bp[1];
    const float* rp = rfo + ((long)(b * UD + j) * ADIM + a) * 2;
    float rr = rp[0], ri = rp[1];
    float mag = sqrtf(rr * rr + ri * ri) + 1e-12f;
    rr /= mag; ri /= mag;
    Fr += br * rr - bi * ri;
    Fi += br * ri + bi * rr;
  }
  __shared__ float red[64];
  red[a] = Fr * Fr + Fi * Fi;
  __syncthreads();
  for (int s = 32; s > 0; s >>= 1) {
    if (a < s) red[a] += red[a + s];
    __syncthreads();
  }
  float nrm = sqrtf(red[0]) + 1e-12f;
  long o = ((long)(b * UD + u) * ADIM + a) * 2;
  out[o + 0] = amp * Fr / nrm;
  out[o + 1] = amp * Fi / nrm;
}

// ----------------------------------------------------------------------------
// Host orchestration
// ----------------------------------------------------------------------------
struct GNp  { const float *a, *b, *g; };
struct RelP { const float *W_edge, *We, *Wk, *Wq, *Wv, *att, *b_edge; };
struct MlpP { const float *b1, *b2; GNp gn1, gn2; const float *w1, *w2; };
struct EmbP { const float *b; GNp gn; const float *w; };
struct LayP { GNp ebn[3]; GNp nbn_ant, nbn_user; RelP rel[3]; };  // a2u,u2a,u2u

extern "C" void kernel_launch(void* const* d_in, const int* in_sizes, int n_in,
                              void* d_out, int out_size, void* d_ws,
                              size_t ws_size, hipStream_t stream) {
  (void)in_sizes; (void)n_in; (void)out_size; (void)ws_size;

  const float* x_user = (const float*)d_in[0];  // [128,4]
  const float* x_ant  = (const float*)d_in[1];  // [256,4]
  // rel index order: 0=a2u, 1=u2a, 2=u2u
  const float* ea_in[3] = {(const float*)d_in[4], (const float*)d_in[3],
                           (const float*)d_in[2]};
  const int* ei[3] = {(const int*)d_in[7], (const int*)d_in[6],
                      (const int*)d_in[5]};

  // params: jax pytree flatten order (keys sorted alphabetically)
  int pi = 8;
  auto FP = [&]() { return (const float*)d_in[pi++]; };
  auto getGN  = [&](GNp& g)  { g.a = FP(); g.b = FP(); g.g = FP(); };
  auto getMLP = [&](MlpP& m) { m.b1 = FP(); m.b2 = FP(); getGN(m.gn1);
                               getGN(m.gn2); m.w1 = FP(); m.w2 = FP(); };
  auto getRel = [&](RelP& r) { r.W_edge = FP(); r.We = FP(); r.Wk = FP();
                               r.Wq = FP(); r.Wv = FP(); r.att = FP();
                               r.b_edge = FP(); };
  MlpP bbmlp; getMLP(bbmlp);
  const float* bbout_b = FP(); const float* bbout_w = FP();
  EmbP emb_ant;  emb_ant.b  = FP(); getGN(emb_ant.gn);  emb_ant.w  = FP();
  EmbP emb_user; emb_user.b = FP(); getGN(emb_user.gn); emb_user.w = FP();
  LayP L[3];
  for (int li = 0; li < 3; ++li) {
    getGN(L[li].ebn[0]); getGN(L[li].ebn[1]); getGN(L[li].ebn[2]);
    getGN(L[li].nbn_ant); getGN(L[li].nbn_user);
    getRel(L[li].rel[0]); getRel(L[li].rel[1]); getRel(L[li].rel[2]);
  }
  MlpP pmlp; getMLP(pmlp);
  const float* pout_b = FP(); const float* pout_w = FP();
  MlpP rfmlp; getMLP(rfmlp);
  const float* rfout_b = FP(); const float* rfout_w = FP();

  // workspace bump allocator
  char* wsp = (char*)d_ws;
  auto alloc = [&](size_t nf) { float* p = (float*)wsp;
                                wsp += nf * sizeof(float); return p; };
  float* xuA = alloc((size_t)128 * 5120); float* xuB = alloc((size_t)128 * 5120);
  float* xaA = alloc((size_t)256 * 5120); float* xaB = alloc((size_t)256 * 5120);
  float* eaA[3] = {alloc((size_t)8192 * 1024), alloc((size_t)8192 * 1024),
                   alloc((size_t)4096 * 1024)};
  float* eaB[3] = {alloc((size_t)8192 * 1024), alloc((size_t)8192 * 1024),
                   alloc((size_t)4096 * 1024)};
  float* qbuf  = alloc((size_t)256 * 5120);
  float* kbuf  = alloc((size_t)256 * 5120);
  float* vbuf  = alloc((size_t)256 * 5120);
  float* eproj = alloc((size_t)8192 * 5120);
  float* logit = alloc((size_t)8192 * HHEADS);
  float* alpha = alloc((size_t)8192 * HHEADS);
  float* ux1 = alloc((size_t)128 * 1024); float* ux2 = alloc((size_t)128 * 512);
  float* Pbuf = alloc(128);
  float* bb1 = alloc((size_t)4096 * 512); float* bb2 = alloc((size_t)4096 * 256);
  float* bbo = alloc((size_t)4096 * 2);
  float* rfsum = alloc((size_t)8192 * 1024);
  float* rf1 = alloc((size_t)8192 * 512); float* rf2 = alloc((size_t)8192 * 256);
  float* rfo = alloc((size_t)8192 * 2);

  auto gemm = [&](const float* Ap, const int* aidx, const float* Bp,
                  const float* bias, float* Cp, int M, int N, int K, int lda,
                  int acc) {
    int tiles = (M >> 4) * (N >> 5);  // N must be a multiple of 32 (true here)
    int blocks = (tiles + 7) / 8;
    gemm_wmma_kernel<<<blocks, 256, 0, stream>>>(Ap, aidx, Bp, bias, Cp, M, N,
                                                 K, lda, acc);
  };
  auto gnorm = [&](float* x, int N, int C, const GNp& g, int act) {
    graph_norm_kernel<<<(C + 255) / 256, 256, 0, stream>>>(x, N, C, g.g, g.b,
                                                           g.a, act);
  };
  auto fill = [&](float* p, long n) {
    fill_kernel<<<(int)((n + 255) / 256), 256, 0, stream>>>(p, n, 0.f);
  };

  // embedding
  embed_kernel<<<(128 * 128 + 255) / 256, 256, 0, stream>>>(
      x_user, 128, 4, emb_user.w, emb_user.b, xuA);
  gnorm(xuA, 128, 128, emb_user.gn, 0);
  embed_kernel<<<(256 * 128 + 255) / 256, 256, 0, stream>>>(
      x_ant, 256, 4, emb_ant.w, emb_ant.b, xaA);
  gnorm(xaA, 256, 128, emb_ant.gn, 0);

  // per-relation dense-edge structure (order: a2u, u2a, u2u)
  int Earr[3]  = {8192, 8192, 4096};
  int deg[3]   = {64, 32, 32};
  int strd[3]  = {1, 64, 32};
  int perB[3]  = {32, 64, 32};
  int epb[3]   = {2048, 2048, 1024};
  int bmul[3]  = {64, 1, 1};
  int NdstA[3] = {128, 256, 128};
  int NsrcA[3] = {256, 128, 128};

  struct { int cin, dh, eout; } cfg[3] = {{128, 32, 256},
                                          {1280, 64, 512},
                                          {2560, 128, 1024}};
  const float* ea_cur[3] = {ea_in[0], ea_in[1], ea_in[2]};
  int ein_cur[3] = {2, 2, 6};
  float* xu_cur = xuA; float* xu_nxt = xuB;
  float* xa_cur = xaA; float* xa_nxt = xaB;

  for (int li = 0; li < 3; ++li) {
    int cin = cfg[li].cin, dh = cfg[li].dh, eout = cfg[li].eout;
    int hd = HHEADS * dh;
    float** ea_nxt = (li % 2 == 0) ? eaA : eaB;
    fill(xu_nxt, (long)128 * hd);
    fill(xa_nxt, (long)256 * hd);

    for (int r = 0; r < 3; ++r) {
      const RelP& rp = L[li].rel[r];
      const float* x_st = (r == 0) ? xa_cur : xu_cur;
      const float* x_dt = (r == 1) ? xa_cur : xu_cur;
      float* outdt      = (r == 1) ? xa_nxt : xu_nxt;
      int Nst = NsrcA[r], Ndt = NdstA[r], E = Earr[r], ein = ein_cur[r];
      const int* src = ei[r];
      const int* dst = ei[r] + E;

      gemm(x_dt, nullptr, rp.Wq, nullptr, qbuf, Ndt, hd, cin, cin, 0);
      gemm(x_st, nullptr, rp.Wk, nullptr, kbuf, Nst, hd, cin, cin, 0);
      gemm(x_st, nullptr, rp.Wv, nullptr, vbuf, Nst, hd, cin, cin, 0);
      gemm(ea_cur[r], nullptr, rp.We, nullptr, eproj, E, hd, ein, ein, 0);

      edge_logit_kernel<<<(E * HHEADS + 255) / 256, 256, 0, stream>>>(
          qbuf, kbuf, eproj, src, dst, rp.att, logit, E, dh, hd);
      seg_softmax_kernel<<<(NdstA[r] * HHEADS + 255) / 256, 256, 0, stream>>>(
          logit, alpha, Ndt, deg[r], strd[r], perB[r], epb[r], bmul[r]);
      aggregate_kernel<<<(int)(((long)Ndt * hd + 255) / 256), 256, 0, stream>>>(
          vbuf, eproj, alpha, src, outdt, Ndt, hd, dh, deg[r], strd[r],
          perB[r], epb[r], bmul[r]);

      // new_ea = [x_st[src] | x_dt[dst] | ea] @ W_edge + b, via 3 gathered GEMMs
      gemm(x_st, src, rp.W_edge, nullptr, ea_nxt[r], E, eout, cin, cin, 0);
      gemm(x_dt, dst, rp.W_edge + (size_t)cin * eout, nullptr, ea_nxt[r], E,
           eout, cin, cin, 1);
      gemm(ea_cur[r], nullptr, rp.W_edge + (size_t)2 * cin * eout, rp.b_edge,
           ea_nxt[r], E, eout, ein, ein, 1);
    }

    gnorm(xu_nxt, 128, hd, L[li].nbn_user, 2);  // elu
    gnorm(xa_nxt, 256, hd, L[li].nbn_ant, 2);
    for (int r = 0; r < 3; ++r) {
      gnorm(ea_nxt[r], Earr[r], eout, L[li].ebn[r], 2);
      ea_cur[r] = ea_nxt[r];
      ein_cur[r] = eout;
    }
    float* t = xu_cur; xu_cur = xu_nxt; xu_nxt = t;
    t = xa_cur; xa_cur = xa_nxt; xa_nxt = t;
  }

  // power head
  gemm(xu_cur, nullptr, pmlp.w1, pmlp.b1, ux1, 128, 1024, 5120, 5120, 0);
  gnorm(ux1, 128, 1024, pmlp.gn1, 1);
  gemm(ux1, nullptr, pmlp.w2, pmlp.b2, ux2, 128, 512, 1024, 1024, 0);
  gnorm(ux2, 128, 512, pmlp.gn2, 1);
  gemv_small_kernel<<<(128 + 255) / 256, 256, 0, stream>>>(ux2, pout_w, pout_b,
                                                           Pbuf, 128, 1, 512);

  // BB head from ea_u2u
  gemm(ea_cur[2], nullptr, bbmlp.w1, bbmlp.b1, bb1, 4096, 512, 1024, 1024, 0);
  gnorm(bb1, 4096, 512, bbmlp.gn1, 1);
  gemm(bb1, nullptr, bbmlp.w2, bbmlp.b2, bb2, 4096, 256, 512, 512, 0);
  gnorm(bb2, 4096, 256, bbmlp.gn2, 1);
  gemv_small_kernel<<<(4096 * 2 + 255) / 256, 256, 0, stream>>>(
      bb2, bbout_w, bbout_b, bbo, 4096, 2, 256);

  // RF head from ea_u2a + ea_a2u
  add_kernel<<<(int)(((long)8192 * 1024 + 255) / 256), 256, 0, stream>>>(
      ea_cur[1], ea_cur[0], rfsum, (long)8192 * 1024);
  gemm(rfsum, nullptr, rfmlp.w1, rfmlp.b1, rf1, 8192, 512, 1024, 1024, 0);
  gnorm(rf1, 8192, 512, rfmlp.gn1, 1);
  gemm(rf1, nullptr, rfmlp.w2, rfmlp.b2, rf2, 8192, 256, 512, 512, 0);
  gnorm(rf2, 8192, 256, rfmlp.gn2, 1);
  gemv_small_kernel<<<(8192 * 2 + 255) / 256, 256, 0, stream>>>(
      rf2, rfout_w, rfout_b, rfo, 8192, 2, 256);

  // final precoder -> d_out [B,U,A,2]
  precoder_kernel<<<BD * UD, ADIM, 0, stream>>>(Pbuf, bbo, rfo, (float*)d_out);
}
